// AttnBlock_48206712930610
// MI455X (gfx1250) — compile-verified, matrix operand order checked by reading, not run on hardware
//
#include <hip/hip_runtime.h>
#include <hip/hip_bf16.h>

// ---------------------------------------------------------------------------
// AttnBlock for MI455X (gfx1250): GroupNorm -> QKV (WMMA bf16 GEMM) ->
// flash attention (WMMA bf16, f32 accum, online softmax, async-LDS Q staging)
// -> proj + residual.  wave32; V_WMMA_F32_16X16X32_BF16 for all matrix math.
// ---------------------------------------------------------------------------

typedef __bf16 bf16x16 __attribute__((ext_vector_type(16)));
typedef float  f32x8   __attribute__((ext_vector_type(8)));
typedef int    v4i     __attribute__((ext_vector_type(4)));

#define B_  4
#define C_  512
#define L_  4096

#if __has_builtin(__builtin_amdgcn_global_load_async_to_lds_b128)
#define HAVE_ASYNC_LDS 1
typedef __attribute__((address_space(1))) v4i gv4i;   // global int4
typedef __attribute__((address_space(3))) v4i lv4i;   // LDS int4
#endif

static __device__ __forceinline__ void wait_asynccnt0() {
#if __has_builtin(__builtin_amdgcn_s_wait_asynccnt)
    __builtin_amdgcn_s_wait_asynccnt(0);
#else
    asm volatile("s_wait_asynccnt 0" ::: "memory");
#endif
}

static __device__ __forceinline__ unsigned short f32_to_bf16(float f) {
    unsigned int u = __float_as_uint(f);
    u += 0x7FFFu + ((u >> 16) & 1u);           // round-to-nearest-even
    return (unsigned short)(u >> 16);
}

union FragU {
    bf16x16 v;
    uint4   q[2];
};

// Load one 16x32 bf16 fragment (A-style rows or B-style columns).
// base points at row 0 of the 16-row group; ld = elements per row; kbase = K offset.
// Per ISA 16-bit layout: lanes 0-15 hold K = kbase+[0..7] and kbase+[16..23];
// lanes 16-31 hold K = kbase+[8..15] and kbase+[24..31].
static __device__ __forceinline__ bf16x16
load_frag(const unsigned short* base, int ld, int kbase, int lane) {
    int m  = lane & 15;
    int kb = (lane & 16) ? 8 : 0;
    const unsigned short* p = base + (size_t)m * ld + kbase + kb;
    FragU f;
    f.q[0] = *(const uint4*)(p);
    f.q[1] = *(const uint4*)(p + 16);
    return f.v;
}

static __device__ __forceinline__ f32x8
wmma_bf16(bf16x16 a, bf16x16 b, f32x8 c) {
    return __builtin_amdgcn_wmma_f32_16x16x32_bf16(
        /*neg_a=*/false, a, /*neg_b=*/false, b,
        /*c_mod=*/(short)0, c, /*reuse_a=*/false, /*reuse_b=*/false);
}

// ---------------------------------------------------------------------------
// Kernel 1: fp32 -> bf16 weight conversion (4 weights at once)
// ---------------------------------------------------------------------------
__global__ void cvt_w_kernel(const float* __restrict__ w0, const float* __restrict__ w1,
                             const float* __restrict__ w2, const float* __restrict__ w3,
                             unsigned short* __restrict__ o0, unsigned short* __restrict__ o1,
                             unsigned short* __restrict__ o2, unsigned short* __restrict__ o3,
                             int n) {
    for (int i = blockIdx.x * blockDim.x + threadIdx.x; i < n; i += gridDim.x * blockDim.x) {
        o0[i] = f32_to_bf16(w0[i]);
        o1[i] = f32_to_bf16(w1[i]);
        o2[i] = f32_to_bf16(w2[i]);
        o3[i] = f32_to_bf16(w3[i]);
    }
}

// ---------------------------------------------------------------------------
// Kernel 2: GroupNorm(32 groups) over x[b,c,l], writes h transposed [b,l,c] bf16
// One block per (batch, group): reduce over 16*4096 elements.
// ---------------------------------------------------------------------------
__global__ void gn_kernel(const float* __restrict__ x,
                          const float* __restrict__ scale,
                          const float* __restrict__ bias,
                          unsigned short* __restrict__ h_t) {
    const int CG = 16;                       // channels per group
    int g  = blockIdx.x & 31;
    int bb = blockIdx.x >> 5;
    const float* xg = x + ((size_t)bb * C_ + (size_t)g * CG) * L_;
    const int N = CG * L_;

    float s = 0.f, ss = 0.f;
    for (int i = threadIdx.x; i < N; i += blockDim.x) {
        float v = xg[i];
        s += v; ss += v * v;
    }
    __shared__ float rs[256], rss[256];
    rs[threadIdx.x] = s; rss[threadIdx.x] = ss;
    __syncthreads();
    for (int o = 128; o > 0; o >>= 1) {
        if ((int)threadIdx.x < o) {
            rs[threadIdx.x]  += rs[threadIdx.x + o];
            rss[threadIdx.x] += rss[threadIdx.x + o];
        }
        __syncthreads();
    }
    float mean = rs[0] / (float)N;
    float var  = rss[0] / (float)N - mean * mean;
    float rstd = rsqrtf(var + 1e-6f);

    for (int i = threadIdx.x; i < N; i += blockDim.x) {
        int ch = i >> 12;                    // i / L_
        int pos = i & (L_ - 1);
        int c = g * CG + ch;
        float v = (xg[i] - mean) * rstd * scale[c] + bias[c];
        h_t[((size_t)bb * L_ + pos) * C_ + c] = f32_to_bf16(v);
    }
}

// ---------------------------------------------------------------------------
// Kernel 3: generic WMMA GEMM  D[m,n] = sum_k A[m,k] * Bt[n,k]  (+bias)(+resid)
// A: bf16 [M,K] row-major (lda), Bt: bf16 [N,K] row-major (ldb), K multiple of 32.
// 128 threads = 4 waves; block tile 64(M) x 64(N); wave w owns rows w*16..w*16+15.
// ---------------------------------------------------------------------------
__global__ void gemm_bt_kernel(const unsigned short* __restrict__ A, long long a_bs, int lda,
                               const unsigned short* __restrict__ Bt, long long b_bs, int ldb,
                               const float* __restrict__ bias, int bias_per_row,
                               void* __restrict__ outv, long long o_bs, int ldo, int out_f32,
                               const float* __restrict__ resid, long long r_bs,
                               int K) {
    int z    = blockIdx.z;
    int n0   = blockIdx.x * 64;
    int m0   = blockIdx.y * 64;
    int wave = threadIdx.x >> 5;
    int lane = threadIdx.x & 31;

    const unsigned short* Ab  = A  + (size_t)z * a_bs + (size_t)(m0 + wave * 16) * lda;
    const unsigned short* Btb = Bt + (size_t)z * b_bs + (size_t)n0 * ldb;

    f32x8 acc[4];
    for (int nf = 0; nf < 4; ++nf) acc[nf] = (f32x8)0.f;

    for (int kk = 0; kk < K; kk += 32) {
        __builtin_prefetch(Ab + kk + 32, 0, 3);           // global_prefetch_b8
        bf16x16 a = load_frag(Ab, lda, kk, lane);
#pragma unroll
        for (int nf = 0; nf < 4; ++nf) {
            bf16x16 b = load_frag(Btb + (size_t)(nf * 16) * ldb, ldb, kk, lane);
            acc[nf] = wmma_bf16(a, b, acc[nf]);
        }
    }

    int col   = lane & 15;
    int rbase = (lane & 16) ? 8 : 0;
#pragma unroll
    for (int nf = 0; nf < 4; ++nf) {
#pragma unroll
        for (int r = 0; r < 8; ++r) {
            int mm = m0 + wave * 16 + rbase + r;
            int nn = n0 + nf * 16 + col;
            float v = acc[nf][r];
            if (bias) v += bias_per_row ? bias[mm] : bias[nn];
            size_t idx = (size_t)z * o_bs + (size_t)mm * ldo + nn;
            if (out_f32) {
                float rv = resid ? resid[(size_t)z * r_bs + (size_t)mm * ldo + nn] : 0.f;
                ((float*)outv)[idx] = v + rv;
            } else {
                ((unsigned short*)outv)[idx] = f32_to_bf16(v);
            }
        }
    }
}

// ---------------------------------------------------------------------------
// Kernel 4: flash attention.
// q_t, k_t: bf16 [b, l, c] (position-major); v: bf16 [b, c, l] (channel-major).
// Block: 256 threads = 8 waves, 32 query rows per block, stream j in tiles of 64.
// Q tile (32x512 bf16 = 32KB) staged once into LDS via async-to-LDS; K/V are
// streamed from global (L2-resident) straight into WMMA B fragments.
// Scores phase: wave w -> fragment (i_sub = w&1, j_sub = w>>1) of S[32x64].
// O phase:      wave w -> rows i_sub = w&1, channels c0 = (w>>1)*128 (8 frags).
// Online softmax per row kept by threads 0..31 (wave 0), exchanged via LDS.
// ---------------------------------------------------------------------------
__global__ void flash_kernel(const unsigned short* __restrict__ q_t,
                             const unsigned short* __restrict__ k_t,
                             const unsigned short* __restrict__ v,
                             unsigned short* __restrict__ o_t,
                             float scale) {
    int b    = blockIdx.y;
    int i0   = blockIdx.x * 32;
    int wave = threadIdx.x >> 5;
    int lane = threadIdx.x & 31;
    int i_sub = wave & 1;
    int j_sub = wave >> 1;                     // 0..3 (scores phase)
    int c0    = (wave >> 1) * 128;             // O phase channel base

    __shared__ unsigned short Qs[32 * C_];     // 32KB staged Q tile [32][512]
    __shared__ float          Sb[32 * 64];
    __shared__ unsigned short Pb[32 * 64];
    __shared__ float          alpha_s[32];
    __shared__ float          linv_s[32];

    // ---- stage Q tile into LDS (async-to-LDS on gfx1250)
    {
        const unsigned short* qsrc = q_t + ((size_t)b * L_ + i0) * C_;
        for (int t = threadIdx.x; t < (32 * C_) / 8; t += blockDim.x) {
#ifdef HAVE_ASYNC_LDS
            __builtin_amdgcn_global_load_async_to_lds_b128(
                (gv4i*)(qsrc + t * 8), (lv4i*)(&Qs[t * 8]),
                /*offset=*/0, /*cpol=*/0);
#else
            *(uint4*)&Qs[t * 8] = *(const uint4*)(qsrc + t * 8);
#endif
        }
#ifdef HAVE_ASYNC_LDS
        wait_asynccnt0();
#endif
        __syncthreads();
    }

    f32x8 Oacc[8];
#pragma unroll
    for (int f = 0; f < 8; ++f) Oacc[f] = (f32x8)0.f;

    float m_run = -3.0e38f;
    float l_run = 0.f;

    const unsigned short* qrow = &Qs[(i_sub * 16) * C_];   // LDS-resident Q rows
    int col   = lane & 15;
    int rbase = (lane & 16) ? 8 : 0;

    for (int j0 = 0; j0 < L_; j0 += 64) {
        // ---- phase A: S[32x64] = scale * Q Kt^T (each wave: one 16x16 frag, K=512)
        f32x8 sacc = (f32x8)0.f;
        const unsigned short* krow = k_t + ((size_t)b * L_ + j0 + j_sub * 16) * C_;
#pragma unroll 4
        for (int kk = 0; kk < C_; kk += 32) {
            bf16x16 a = load_frag(qrow, C_, kk, lane);     // ds_load_b128
            bf16x16 bb = load_frag(krow, C_, kk, lane);    // global_load_b128
            sacc = wmma_bf16(a, bb, sacc);
        }
#pragma unroll
        for (int r = 0; r < 8; ++r)
            Sb[(i_sub * 16 + rbase + r) * 64 + j_sub * 16 + col] = sacc[r] * scale;
        __syncthreads();

        // ---- phase B: online softmax update (wave 0, one thread per row)
        if (threadIdx.x < 32) {
            int row = threadIdx.x;
            float mt = -3.0e38f;
            for (int jj = 0; jj < 64; ++jj) mt = fmaxf(mt, Sb[row * 64 + jj]);
            float mnew = fmaxf(m_run, mt);
            float al   = __expf(m_run - mnew);      // 0 on first tile
            float rsum = 0.f;
            for (int jj = 0; jj < 64; ++jj) {
                float p = __expf(Sb[row * 64 + jj] - mnew);
                rsum += p;
                Pb[row * 64 + jj] = f32_to_bf16(p);
            }
            l_run = l_run * al + rsum;
            m_run = mnew;
            alpha_s[row] = al;
        }
        __syncthreads();

        // ---- phase C: O = diag(alpha)*O + P * V^T (each wave: 8 channel frags)
        float av[8];
#pragma unroll
        for (int r = 0; r < 8; ++r) av[r] = alpha_s[i_sub * 16 + rbase + r];
#pragma unroll
        for (int f = 0; f < 8; ++f)
#pragma unroll
            for (int r = 0; r < 8; ++r) Oacc[f][r] *= av[r];

#pragma unroll
        for (int ks = 0; ks < 2; ++ks) {
            int kb2 = ks * 32;
            bf16x16 a = load_frag(&Pb[(i_sub * 16) * 64], 64, kb2, lane);  // ds_load
#pragma unroll
            for (int f = 0; f < 8; ++f) {
                const unsigned short* vrow = v + ((size_t)b * C_ + c0 + f * 16) * L_ + j0;
                bf16x16 bb = load_frag(vrow, L_, kb2, lane);
                Oacc[f] = wmma_bf16(a, bb, Oacc[f]);
            }
        }
        __syncthreads();
    }

    if (threadIdx.x < 32) linv_s[threadIdx.x] = 1.0f / l_run;
    __syncthreads();

    float lv[8];
#pragma unroll
    for (int r = 0; r < 8; ++r) lv[r] = linv_s[i_sub * 16 + rbase + r];
#pragma unroll
    for (int f = 0; f < 8; ++f)
#pragma unroll
        for (int r = 0; r < 8; ++r) {
            size_t row = (size_t)b * L_ + i0 + i_sub * 16 + rbase + r;
            o_t[row * C_ + c0 + f * 16 + col] = f32_to_bf16(Oacc[f][r] * lv[r]);
        }
}

// ---------------------------------------------------------------------------
extern "C" void kernel_launch(void* const* d_in, const int* in_sizes, int n_in,
                              void* d_out, int out_size, void* d_ws, size_t ws_size,
                              hipStream_t stream) {
    const float* x        = (const float*)d_in[0];
    const float* gn_scale = (const float*)d_in[1];
    const float* gn_bias  = (const float*)d_in[2];
    const float* wq = (const float*)d_in[3];  const float* bq = (const float*)d_in[4];
    const float* wk = (const float*)d_in[5];  const float* bk = (const float*)d_in[6];
    const float* wv = (const float*)d_in[7];  const float* bv = (const float*)d_in[8];
    const float* wp = (const float*)d_in[9];  const float* bp = (const float*)d_in[10];

    const size_t WN  = (size_t)C_ * C_;        // 262144
    const size_t TN  = (size_t)L_ * C_;        // per-batch activation elems
    unsigned short* wq_b = (unsigned short*)d_ws;
    unsigned short* wk_b = wq_b + WN;
    unsigned short* wv_b = wk_b + WN;
    unsigned short* wp_b = wv_b + WN;
    unsigned short* h_t  = wp_b + WN;          // [b, l, c]
    unsigned short* q_t  = h_t + (size_t)B_ * TN;
    unsigned short* k_t  = q_t + (size_t)B_ * TN;
    unsigned short* v_m  = k_t + (size_t)B_ * TN;   // [b, c, l]
    unsigned short* o_t  = v_m + (size_t)B_ * TN;   // [b, l, c]

    // 1) weights -> bf16
    cvt_w_kernel<<<512, 256, 0, stream>>>(wq, wk, wv, wp, wq_b, wk_b, wv_b, wp_b, (int)WN);

    // 2) GroupNorm -> h_t [b,l,c] bf16
    gn_kernel<<<B_ * 32, 256, 0, stream>>>(x, gn_scale, gn_bias, h_t);

    // 3) q_t[i,o] = h_t[i,:].Wq[o,:] + bq[o]   (M=L, N=C)
    dim3 gQ(C_ / 64, L_ / 64, B_);
    gemm_bt_kernel<<<gQ, 128, 0, stream>>>(h_t, (long long)TN, C_,
                                           wq_b, 0LL, C_,
                                           bq, /*bias_per_row=*/0,
                                           q_t, (long long)TN, C_, /*out_f32=*/0,
                                           nullptr, 0LL, C_);
    gemm_bt_kernel<<<gQ, 128, 0, stream>>>(h_t, (long long)TN, C_,
                                           wk_b, 0LL, C_,
                                           bk, 0,
                                           k_t, (long long)TN, C_, 0,
                                           nullptr, 0LL, C_);

    // 4) v[o,i] = Wv[o,:].h_t[i,:] + bv[o]    (M=C, N=L), channel-major out
    dim3 gV(L_ / 64, C_ / 64, B_);
    gemm_bt_kernel<<<gV, 128, 0, stream>>>(wv_b, 0LL, C_,
                                           h_t, (long long)TN, C_,
                                           bv, /*bias_per_row=*/1,
                                           v_m, (long long)TN, L_, 0,
                                           nullptr, 0LL, C_);

    // 5) flash attention -> o_t [b,l,c]
    dim3 gF(L_ / 32, B_);
    const float scale = 0.04419417382415922f;   // 512^-0.5
    flash_kernel<<<gF, 256, 0, stream>>>(q_t, k_t, v_m, o_t, scale);

    // 6) out[o,i] = x[o,i] + bp[o] + Wp[o,:].o_t[i,:]
    gemm_bt_kernel<<<gV, 128, 0, stream>>>(wp_b, 0LL, C_,
                                           o_t, (long long)TN, C_,
                                           bp, 1,
                                           d_out, (long long)TN, L_, /*out_f32=*/1,
                                           x, (long long)TN, C_);
}
